// AnyPrecisionLinear_57767310131825
// MI455X (gfx1250) — matrix operand order repaired and checked by reading, not compiled, not run on hardware
//
#include <hip/hip_runtime.h>
#include <cfloat>
#include <stdint.h>

typedef __attribute__((ext_vector_type(16))) _Float16 v16h;
typedef __attribute__((ext_vector_type(8)))  _Float16 v8h;
typedef __attribute__((ext_vector_type(8)))  float    v8f;

#define WBITS 4
#define BM 128
#define BN 128
#define BK 32
#define LDT 40   // padded LDS row stride in halfs (20 words -> conflict-free frag reads)

// ---------------- 1) bit-plane LUT dequant:  qweight[4,O,I/32] + lut[O,16] -> Wh[O,I] (f16)
__global__ void apl_dequant_kernel(const int* __restrict__ qw,
                                   const float* __restrict__ lut,
                                   _Float16* __restrict__ Wh,
                                   int O, int Iw) {
    int gid = blockIdx.x * blockDim.x + threadIdx.x;
    int total = O * Iw;
    if (gid >= total) return;
    int o = gid / Iw;
    int w = gid - o * Iw;
    int p0 = qw[0 * total + o * Iw + w];   // MSB plane -> weight 8
    int p1 = qw[1 * total + o * Iw + w];
    int p2 = qw[2 * total + o * Iw + w];
    int p3 = qw[3 * total + o * Iw + w];   // LSB plane -> weight 1
    float l[16];
#pragma unroll
    for (int i = 0; i < 16; ++i) l[i] = lut[o * 16 + i];
    _Float16* dst = Wh + (size_t)o * ((size_t)Iw * 32) + (size_t)w * 32;
#pragma unroll
    for (int j = 0; j < 32; ++j) {
        int idx = (((p0 >> j) & 1) << 3) | (((p1 >> j) & 1) << 2) |
                  (((p2 >> j) & 1) << 1) | ((p3 >> j) & 1);
        dst[j] = (_Float16)l[idx];
    }
}

// ---------------- 2) CSR outlier scatter-add into Wh (one thread per row -> no atomics)
__global__ void apl_csr_kernel(const int* __restrict__ rows,
                               const int* __restrict__ cols,
                               const float* __restrict__ vals,
                               _Float16* __restrict__ Wh, int O, int I) {
    int o = blockIdx.x * blockDim.x + threadIdx.x;
    if (o >= O) return;
    int k0 = rows[o], k1 = rows[o + 1];
    for (int k = k0; k < k1; ++k) {
        size_t off = (size_t)o * I + (size_t)cols[k];
        Wh[off] = (_Float16)((float)Wh[off] + vals[k]);
    }
}

// ---------------- 3) x fp32 -> f16
__global__ void apl_cvt_kernel(const float* __restrict__ x,
                               _Float16* __restrict__ xh, size_t n4) {
    size_t i = ((size_t)blockIdx.x * blockDim.x + threadIdx.x);
    if (i >= n4) return;
    const float4 v = *(const float4*)(x + i * 4);
    _Float16* d = xh + i * 4;
    d[0] = (_Float16)v.x; d[1] = (_Float16)v.y;
    d[2] = (_Float16)v.z; d[3] = (_Float16)v.w;
}

// ---- CDNA5 async global->LDS copy (ASYNCcnt-tracked, no VGPR round trip) ----
__device__ __forceinline__ void async_copy_b128(uint32_t lds_off, const _Float16* g) {
    asm volatile("global_load_async_to_lds_b128 %0, %1, off"
                 :: "v"(lds_off), "v"((unsigned long long)(uintptr_t)g)
                 : "memory");
}
__device__ __forceinline__ void wait_asynccnt0() {
    asm volatile("s_wait_asynccnt 0x0" ::: "memory");
}

// ---------------- 4) WMMA GEMM: out[M,N] = clamp(Xh[M,K] * Wh[N,K]^T + bias)
// Double-buffered LDS; async loads for tile t+1 overlap WMMA compute on tile t.
__global__ __launch_bounds__(256)
void apl_wmma_gemm_kernel(const _Float16* __restrict__ Xh,
                          const _Float16* __restrict__ Wh,
                          const float* __restrict__ bias,
                          float* __restrict__ out,
                          int M, int N, int K, float clampv) {
    __shared__ __align__(16) _Float16 lA[2][BM * LDT];
    __shared__ __align__(16) _Float16 lB[2][BN * LDT];

    const int tid  = threadIdx.x;
    const int lane = tid & 31;
    const int wid  = tid >> 5;       // 0..7
    const int wm   = wid >> 2;       // 0..1 -> 64 rows per wave
    const int wn   = wid & 3;        // 0..3 -> 32 cols per wave
    const int hsel = lane >> 4;      // upper/lower half of wave
    const int lN   = lane & 15;

    const int m0 = blockIdx.y * BM;
    const int n0 = blockIdx.x * BN;

    v8f acc[4][2];
#pragma unroll
    for (int i = 0; i < 4; ++i)
#pragma unroll
        for (int j = 0; j < 2; ++j)
#pragma unroll
            for (int r = 0; r < 8; ++r) acc[i][j][r] = 0.0f;

    // cooperative tile loads: 256 threads x 4 async b128 per K-step
    const int lr0 = tid >> 2;            // row 0..63; +64 for second half
    const int lch = (tid & 3) * 8;       // half-offset 0,8,16,24
    const _Float16* gA0 = Xh + (size_t)(m0 + lr0) * K + lch;
    const _Float16* gA1 = Xh + (size_t)(m0 + lr0 + 64) * K + lch;
    const _Float16* gB0 = Wh + (size_t)(n0 + lr0) * K + lch;
    const _Float16* gB1 = Wh + (size_t)(n0 + lr0 + 64) * K + lch;

    // LDS byte offsets (low 32 bits of generic pointer = LDS address)
    uint32_t dA0[2], dA1[2], dB0[2], dB1[2];
#pragma unroll
    for (int b = 0; b < 2; ++b) {
        dA0[b] = (uint32_t)(uintptr_t)&lA[b][lr0 * LDT + lch];
        dA1[b] = (uint32_t)(uintptr_t)&lA[b][(lr0 + 64) * LDT + lch];
        dB0[b] = (uint32_t)(uintptr_t)&lB[b][lr0 * LDT + lch];
        dB1[b] = (uint32_t)(uintptr_t)&lB[b][(lr0 + 64) * LDT + lch];
    }

    // prologue: async-fill buffer 0 with tile 0
    async_copy_b128(dA0[0], gA0);
    async_copy_b128(dA1[0], gA1);
    async_copy_b128(dB0[0], gB0);
    async_copy_b128(dB1[0], gB1);
    wait_asynccnt0();
    __syncthreads();

    const int T = K / BK;
    for (int t = 0; t < T; ++t) {
        const int buf = t & 1;
        if (t + 1 < T) {  // issue next tile's async loads; overlap with WMMA below
            const int kn = (t + 1) * BK;
            async_copy_b128(dA0[buf ^ 1], gA0 + kn);
            async_copy_b128(dA1[buf ^ 1], gA1 + kn);
            async_copy_b128(dB0[buf ^ 1], gB0 + kn);
            async_copy_b128(dB1[buf ^ 1], gB1 + kn);
        }

        const _Float16* A = lA[buf];
        const _Float16* B = lB[buf];

        // A fragments: lane holds row M=lN; K chunks at c0 and c0+16, c0 = 8*hsel
        union AF { v16h v; v8h h[2]; } af[4];
        const int c0 = hsel * 8;
#pragma unroll
        for (int i = 0; i < 4; ++i) {
            const int am = wm * 64 + i * 16 + lN;
            af[i].h[0] = *(const v8h*)(A + am * LDT + c0);
            af[i].h[1] = *(const v8h*)(A + am * LDT + c0 + 16);
        }
#pragma unroll
        for (int j = 0; j < 2; ++j) {
            // B fragment: lane holds col N=lN; K chunks at kb and kb+16, kb = 8*hsel
            union BF { v16h v; v8h h[2]; } bf;
            const int bn = wn * 32 + j * 16 + lN;
            const int kb = hsel * 8;
            bf.h[0] = *(const v8h*)(B + bn * LDT + kb);
            bf.h[1] = *(const v8h*)(B + bn * LDT + kb + 16);
#pragma unroll
            for (int i = 0; i < 4; ++i) {
                acc[i][j] = __builtin_amdgcn_wmma_f32_16x16x32_f16(
                    false, af[i].v, false, bf.v,
                    (short)0, acc[i][j], false, false);
            }
        }

        if (t + 1 < T) {
            wait_asynccnt0();
            __syncthreads();
        }
    }

    // epilogue: C/D layout -> VGPR r: M = r + 8*hsel, N = lN
#pragma unroll
    for (int j = 0; j < 2; ++j) {
        const int ncol = n0 + wn * 32 + j * 16 + lN;
        const float bv = bias[ncol];
#pragma unroll
        for (int i = 0; i < 4; ++i) {
#pragma unroll
            for (int r = 0; r < 8; ++r) {
                const int mrow = m0 + wm * 64 + i * 16 + r + hsel * 8;
                float v = acc[i][j][r] + bv;
                v = fminf(fmaxf(v, -clampv), clampv);
                out[(size_t)mrow * N + ncol] = v;
            }
        }
    }
}

extern "C" void kernel_launch(void* const* d_in, const int* in_sizes, int n_in,
                              void* d_out, int out_size, void* d_ws, size_t ws_size,
                              hipStream_t stream) {
    (void)n_in; (void)out_size; (void)ws_size;
    const float* x       = (const float*)d_in[0];
    const int*   qweight = (const int*)d_in[1];
    const float* lut     = (const float*)d_in[2];
    const int*   rows    = (const int*)d_in[3];
    const int*   cols    = (const int*)d_in[4];
    const float* vals    = (const float*)d_in[5];
    const float* bias    = (const float*)d_in[6];
    float*       out     = (float*)d_out;

    const int O  = in_sizes[2] / 16;            // lut [O, 2^4]
    const int Iw = in_sizes[1] / (WBITS * O);   // qweight [4, O, I/32]
    const int I  = Iw * 32;
    const int M  = in_sizes[0] / I;             // B*S tokens

    _Float16* Wh = (_Float16*)d_ws;             // [O, I] dequantized weights
    _Float16* Xh = Wh + (size_t)O * I;          // [M, I] f16 activations

    // 1) dequant
    {
        int total = O * Iw;
        apl_dequant_kernel<<<(total + 255) / 256, 256, 0, stream>>>(qweight, lut, Wh, O, Iw);
    }
    // 2) sparse outliers
    apl_csr_kernel<<<(O + 255) / 256, 256, 0, stream>>>(rows, cols, vals, Wh, O, I);
    // 3) x -> f16
    {
        size_t n4 = ((size_t)M * I) / 4;
        apl_cvt_kernel<<<(unsigned)((n4 + 255) / 256), 256, 0, stream>>>(x, Xh, n4);
    }
    // 4) WMMA GEMM + bias + clamp
    {
        const float clampv = FLT_MAX * (1.0f - 0.005f);
        dim3 grid(O / BN, M / BM);
        apl_wmma_gemm_kernel<<<grid, 256, 0, stream>>>(Xh, Wh, bias, out, M, O, I, clampv);
    }
}
